// ChaosAwareAttention_50190987821059
// MI455X (gfx1250) — matrix operand order repaired and checked by reading, not compiled, not run on hardware
//
#include <hip/hip_runtime.h>

// ---------------------------------------------------------------------------
// ChaosAwareAttention for MI455X (gfx1250, wave32, WMMA + async-LDS).
// B=2 S=2048 D=1024 H=16 hd=64 C=16.
//  * all 4 GEMMs on v_wmma_f32_16x16x32_bf16
//  * QKV / OUT GEMMs: 128x64 block tiles, double-buffered LDS staging filled
//    with global_load_async_to_lds_b128 (ASYNCcnt) -> 8x/4x operand reuse
//  * attention: 32-query-row score strip resident in LDS (256KB of the WGP's
//    320KB), softmax in LDS, 537MB `w` output written exactly once,
//    att = w@v WMMA reads A from LDS (f32->bf16 on the fly) and B from a
//    transposed vT copy with b128 loads.
// ---------------------------------------------------------------------------

typedef __attribute__((ext_vector_type(16))) __bf16 v16bf;
typedef __attribute__((ext_vector_type(8)))  float  v8f;

union Frag {
    v16bf v;
    uint4 q[2];
    unsigned int u[8];
};

__device__ __forceinline__ unsigned short f2bf(float f) {
    unsigned u = __float_as_uint(f);
    return (unsigned short)((u + 0x7FFFu + ((u >> 16) & 1u)) >> 16); // RNE
}
__device__ __forceinline__ unsigned pack_bf16(float lo, float hi) {
    return (unsigned)f2bf(lo) | ((unsigned)f2bf(hi) << 16);
}

// CDNA5 async global->LDS copy (16B), tracked by ASYNCcnt.
__device__ __forceinline__ void async_copy_b128(unsigned lds_off, const void* gptr) {
    asm volatile("global_load_async_to_lds_b128 %0, %1, off"
                 :: "v"(lds_off), "v"(gptr) : "memory");
}
__device__ __forceinline__ void wait_async0() {
    asm volatile("s_wait_asynccnt 0" ::: "memory");
}

// Constants block (floats) in workspace:
#define CH_GA   0
#define CH_GV   16
#define CH_TEMP 32
#define CH_G    33
#define CH_BIAS 48
#define CH_VADD 64

// ---------------------------------------------------------------------------
// Kernel 1: Wqkv [3072,1024] / Wout [1024,1024] -> bf16 ([N,K] row major).
// ---------------------------------------------------------------------------
__global__ void k_cvt_weights(const float* __restrict__ Wqkv,
                              const float* __restrict__ Wout,
                              unsigned short* __restrict__ wq,
                              unsigned short* __restrict__ wo) {
    int stride = gridDim.x * blockDim.x;
    int i = blockIdx.x * blockDim.x + threadIdx.x;
    for (int j = i; j < 3072 * 1024; j += stride) wq[j] = f2bf(Wqkv[j]);
    for (int j = i; j < 1024 * 1024; j += stride) wo[j] = f2bf(Wout[j]);
}

// ---------------------------------------------------------------------------
// Kernel 2: chaos encoder + gates + temp + sc/bias/v_add (tiny, one block).
// ---------------------------------------------------------------------------
__global__ void k_chaos(const float* __restrict__ cf,
                        const float* __restrict__ W1, const float* __restrict__ b1,
                        const float* __restrict__ g1, const float* __restrict__ bb1,
                        const float* __restrict__ W2, const float* __restrict__ b2,
                        const float* __restrict__ g2, const float* __restrict__ bb2,
                        const float* __restrict__ Wga, const float* __restrict__ bga,
                        const float* __restrict__ Wgv, const float* __restrict__ bgv,
                        const float* __restrict__ Wt,  const float* __restrict__ bt,
                        const float* __restrict__ qrot, const float* __restrict__ fscales,
                        float* __restrict__ ch) {
    __shared__ float buf[1024];
    __shared__ float buf2[1024];
    __shared__ float red[256], red2[256];
    __shared__ float shGv[16];
    __shared__ float cfl[16];
    int tid = threadIdx.x;
    if (tid < 16) cfl[tid] = cf[tid];
    __syncthreads();

    for (int d = tid; d < 1024; d += 256) {
        float a = b1[d];
        #pragma unroll
        for (int j = 0; j < 16; ++j) a += W1[d * 16 + j] * cfl[j];
        buf[d] = a;
    }
    __syncthreads();
    float s = 0.f, s2 = 0.f;
    for (int d = tid; d < 1024; d += 256) { float v = buf[d]; s += v; s2 += v * v; }
    red[tid] = s; red2[tid] = s2; __syncthreads();
    for (int o = 128; o > 0; o >>= 1) {
        if (tid < o) { red[tid] += red[tid + o]; red2[tid] += red2[tid + o]; }
        __syncthreads();
    }
    float mean = red[0] / 1024.f;
    float rstd = rsqrtf(red2[0] / 1024.f - mean * mean + 1e-5f);
    __syncthreads();
    for (int d = tid; d < 1024; d += 256) {
        float h = (buf[d] - mean) * rstd * g1[d] + bb1[d];
        buf[d] = h > 0.f ? h : 0.f;
    }
    __syncthreads();
    for (int d = tid; d < 1024; d += 256) {
        float a = b2[d];
        const float* w = W2 + (size_t)d * 1024;
        for (int j = 0; j < 1024; ++j) a += w[j] * buf[j];
        buf2[d] = a;
    }
    __syncthreads();
    s = 0.f; s2 = 0.f;
    for (int d = tid; d < 1024; d += 256) { float v = buf2[d]; s += v; s2 += v * v; }
    red[tid] = s; red2[tid] = s2; __syncthreads();
    for (int o = 128; o > 0; o >>= 1) {
        if (tid < o) { red[tid] += red[tid + o]; red2[tid] += red2[tid + o]; }
        __syncthreads();
    }
    mean = red[0] / 1024.f;
    rstd = rsqrtf(red2[0] / 1024.f - mean * mean + 1e-5f);
    __syncthreads();
    for (int d = tid; d < 1024; d += 256)
        buf2[d] = (buf2[d] - mean) * rstd * g2[d] + bb2[d];   // ce [H,hd]

    if (tid < 16) {
        float a = bga[tid], v = bgv[tid];
        #pragma unroll
        for (int j = 0; j < 16; ++j) {
            a += Wga[tid * 16 + j] * cfl[j];
            v += Wgv[tid * 16 + j] * cfl[j];
        }
        float ga = 1.f / (1.f + __expf(-a));
        float gv = 1.f / (1.f + __expf(-v));
        ch[CH_GA + tid] = ga; ch[CH_GV + tid] = gv; shGv[tid] = gv;
    }
    if (tid == 0) {
        float t = bt[0];
        #pragma unroll
        for (int j = 0; j < 16; ++j) t += Wt[j] * cfl[j];
        ch[CH_TEMP] = __logf(1.f + __expf(t)) + 1.f;
        float m = 0.f;
        #pragma unroll
        for (int j = 0; j < 16; ++j) m += cfl[j];
        m *= (1.f / 16.f);
        ch[CH_G] = (1.f / (1.f + __expf(-m))) * 0.5f + 0.5f;
    }
    __syncthreads();
    float fs = 1.f / (1.f + __expf(-fscales[0]));
    for (int i = tid; i < 1024; i += 256) {
        int h = i >> 6, dd = i & 63;
        float a = 0.f;
        for (int j = 0; j < 64; ++j) a += buf2[h * 64 + j] * qrot[j * 64 + dd];
        a *= fs;
        buf[i] = a;
        ch[CH_VADD + i] = a * shGv[h];
    }
    __syncthreads();
    if (tid < 16) {
        float b = 0.f;
        for (int j = 0; j < 64; ++j) b += buf[tid * 64 + j];
        ch[CH_BIAS + tid] = b;
    }
}

// ---------------------------------------------------------------------------
// Kernel 3: LayerNorm (pre): x[4096,1024] -> xn f32 + bf16 copy.
// ---------------------------------------------------------------------------
__global__ void __launch_bounds__(256)
k_ln_pre(const float* __restrict__ x, const float* __restrict__ g,
         const float* __restrict__ b, float* __restrict__ xn,
         unsigned short* __restrict__ xbf) {
    __shared__ float red[256], red2[256];
    int row = blockIdx.x, tid = threadIdx.x;
    const float* xr = x + (size_t)row * 1024;
    float v[4], s = 0.f, s2 = 0.f;
    #pragma unroll
    for (int i = 0; i < 4; ++i) { v[i] = xr[tid + 256 * i]; s += v[i]; s2 += v[i] * v[i]; }
    red[tid] = s; red2[tid] = s2; __syncthreads();
    for (int o = 128; o > 0; o >>= 1) {
        if (tid < o) { red[tid] += red[tid + o]; red2[tid] += red2[tid + o]; }
        __syncthreads();
    }
    float mean = red[0] / 1024.f;
    float rstd = rsqrtf(red2[0] / 1024.f - mean * mean + 1e-5f);
    #pragma unroll
    for (int i = 0; i < 4; ++i) {
        int c = tid + 256 * i;
        float o = (v[i] - mean) * rstd * g[c] + b[c];
        xn[(size_t)row * 1024 + c]  = o;
        xbf[(size_t)row * 1024 + c] = f2bf(o);
    }
}

// ---------------------------------------------------------------------------
// Kernel 4: QKV GEMM, LDS-tiled. Block = 128(M) x 64(N), 8 waves; wave w owns
// row-subtile w and all 4 col-subtiles (4 accumulators). K staged in LDS in
// 32-wide double-buffered chunks via async b128 copies.
// Writes q/k/v [B,H,S,hd] bf16 (+bias, v += gated chaos v_add) and vT
// [B,H,hd,S] bf16 for the attention w@v pass.
// ---------------------------------------------------------------------------
__global__ void __launch_bounds__(256)
k_qkv_gemm(const unsigned short* __restrict__ A, const unsigned short* __restrict__ W,
           const float* __restrict__ bqkv, const float* __restrict__ ch,
           unsigned short* __restrict__ qb, unsigned short* __restrict__ kb,
           unsigned short* __restrict__ vb, unsigned short* __restrict__ vt) {
    __shared__ unsigned short As[2][128 * 32];   // 8KB x2
    __shared__ unsigned short Bs[2][64 * 32];    // 4KB x2
    int tid = threadIdx.x, lane = tid & 31, wv = tid >> 5, ln = lane & 15;
    int blk = blockIdx.x;
    int mt = blk / 48, nt = blk % 48;            // M/128=32, N/64=48
    int m0 = mt * 128, n0 = nt * 64;
    int khalfA = (lane >> 4) * 8, khalfB = (lane >> 4) * 8, hi8 = khalfA;

    unsigned aoff[2] = { (unsigned)(size_t)&As[0][0], (unsigned)(size_t)&As[1][0] };
    unsigned boff[2] = { (unsigned)(size_t)&Bs[0][0], (unsigned)(size_t)&Bs[1][0] };

    auto stage = [&](int k0, int buf) {
        // A tile: 128 rows x 64B -> 512 x 16B chunks, 2 per thread
        #pragma unroll
        for (int c = tid; c < 512; c += 256) {
            int row = c >> 2, seg = c & 3;
            async_copy_b128(aoff[buf] + (unsigned)(row * 64 + seg * 16),
                            A + (size_t)(m0 + row) * 1024 + k0 + seg * 8);
        }
        // B tile: 64 rows x 64B -> 256 x 16B chunks, 1 per thread
        {
            int row = tid >> 2, seg = tid & 3;
            async_copy_b128(boff[buf] + (unsigned)(tid * 16),
                            W + (size_t)(n0 + row) * 1024 + k0 + seg * 8);
        }
    };

    v8f acc[4] = {};
    stage(0, 0);
    wait_async0();
    __syncthreads();
    for (int ks = 0; ks < 32; ++ks) {
        int buf = ks & 1;
        if (ks + 1 < 32) stage((ks + 1) * 32, buf ^ 1);
        Frag a;
        const unsigned short* ap = &As[buf][(wv * 16 + ln) * 32 + khalfA];
        a.q[0] = *(const uint4*)ap; a.q[1] = *(const uint4*)(ap + 16);
        #pragma unroll
        for (int nb = 0; nb < 4; ++nb) {
            Frag b;
            const unsigned short* bp = &Bs[buf][(nb * 16 + ln) * 32 + khalfB];
            b.q[0] = *(const uint4*)bp; b.q[1] = *(const uint4*)(bp + 16);
            acc[nb] = __builtin_amdgcn_wmma_f32_16x16x32_bf16(false, a.v, false, b.v,
                                                              (short)0, acc[nb], false, false);
        }
        wait_async0();          // prefetch for buf^1 done; issued before compute
        __syncthreads();
    }

    int mr = m0 + wv * 16;
    #pragma unroll
    for (int nb = 0; nb < 4; ++nb) {
        int nc = n0 + nb * 16 + ln;
        int t = nc >> 10;                        // 0=q 1=k 2=v
        int rem = nc & 1023;
        int h = rem >> 6, dd = rem & 63;
        float bias = bqkv[nc];
        float vadd = (t == 2) ? ch[CH_VADD + rem] : 0.f;
        unsigned short* dst = (t == 0) ? qb : (t == 1) ? kb : vb;
        #pragma unroll
        for (int r = 0; r < 8; ++r) {
            int m = mr + r + hi8;                // flattened b*S+s
            int b_ = m >> 11, s_ = m & 2047;
            unsigned short o = f2bf(acc[nb][r] + bias + vadd);
            size_t bh64 = ((size_t)b_ * 16 + h);
            dst[(bh64 * 2048 + s_) * 64 + dd] = o;
            if (t == 2) vt[(bh64 * 64 + dd) * 2048 + s_] = o;   // transposed copy
        }
    }
}

// ---------------------------------------------------------------------------
// Kernel 5: fused attention. Block = (b,h, 32-query strip). 256KB LDS holds
// the full 32x2048 f32 score strip. 8 waves do scores + softmax; waves 0..3
// do att = w @ v with 2 accumulators each (rowblocks 0/1), converting w
// f32->bf16 on the fly for A fragments, B fragments from vT via b128 loads.
// ---------------------------------------------------------------------------
#define ATT_SMEM (32*2048*4 + 256*4 + 32*4)

__global__ void __launch_bounds__(256)
k_attention(const unsigned short* __restrict__ qbuf,
            const unsigned short* __restrict__ kbuf,
            const unsigned short* __restrict__ vt,
            const float* __restrict__ ch,
            float* __restrict__ w_out, unsigned short* __restrict__ att_bf) {
    extern __shared__ char smem[];
    float* scs     = (float*)smem;                         // 32 x 2048 f32
    float* red     = (float*)(smem + 32 * 2048 * 4);       // 32 x 8
    float* rowstat = red + 256;                            // 32

    int tid = threadIdx.x, lane = tid & 31, wv = tid >> 5, ln = lane & 15;
    int blk = blockIdx.x;
    int qt = blk & 63;             // S/32 strips
    int bh = blk >> 6;             // b*H + h
    int h = bh & 15;
    int q0 = qt * 32;
    const unsigned short* qp = qbuf + ((size_t)bh * 2048 + q0) * 64;
    const unsigned short* kp = kbuf + (size_t)bh * 2048 * 64;
    float ascale = ch[CH_GA + h] / (8.f * ch[CH_TEMP]);
    float cbias  = ch[CH_BIAS + h] / ch[CH_TEMP];
    int khalfA = (lane >> 4) * 8, khalfB = (lane >> 4) * 8, hi8 = khalfA;

    // q A-fragments: 2 rowblocks x 2 k-halves
    Frag aq[2][2];
    #pragma unroll
    for (int rb = 0; rb < 2; ++rb) {
        const unsigned short* ap = qp + (size_t)(rb * 16 + ln) * 64 + khalfA;
        aq[rb][0].q[0] = *(const uint4*)(ap);      aq[rb][0].q[1] = *(const uint4*)(ap + 16);
        aq[rb][1].q[0] = *(const uint4*)(ap + 32); aq[rb][1].q[1] = *(const uint4*)(ap + 48);
    }

    // ---- Phase 1: scores -> LDS (f32, with gate/bias/temp folded) ----
    for (int cb = wv; cb < 128; cb += 8) {
        const unsigned short* bp = kp + (size_t)(cb * 16 + ln) * 64 + khalfB;
        Frag b0, b1;
        b0.q[0] = *(const uint4*)(bp);      b0.q[1] = *(const uint4*)(bp + 16);
        b1.q[0] = *(const uint4*)(bp + 32); b1.q[1] = *(const uint4*)(bp + 48);
        #pragma unroll
        for (int rb = 0; rb < 2; ++rb) {
            v8f acc = {};
            acc = __builtin_amdgcn_wmma_f32_16x16x32_bf16(false, aq[rb][0].v, false, b0.v,
                                                          (short)0, acc, false, false);
            acc = __builtin_amdgcn_wmma_f32_16x16x32_bf16(false, aq[rb][1].v, false, b1.v,
                                                          (short)0, acc, false, false);
            #pragma unroll
            for (int r = 0; r < 8; ++r)
                scs[(rb * 16 + r + hi8) * 2048 + cb * 16 + ln] = acc[r] * ascale + cbias;
        }
    }
    __syncthreads();

    // ---- Phase 2: softmax; 8 threads per row, 256 cols each, float4 ----
    int row = tid >> 3, sub = tid & 7;
    float4* sr4 = (float4*)(scs + row * 2048 + sub * 256);
    float mx = -1e30f;
    for (int c = 0; c < 64; ++c) {
        float4 v = sr4[c];
        mx = fmaxf(mx, fmaxf(fmaxf(v.x, v.y), fmaxf(v.z, v.w)));
    }
    red[tid] = mx; __syncthreads();
    if (sub == 0) {
        float m = red[row * 8];
        for (int j = 1; j < 8; ++j) m = fmaxf(m, red[row * 8 + j]);
        rowstat[row] = m;
    }
    __syncthreads();
    float rmax = rowstat[row], sm = 0.f;
    for (int c = 0; c < 64; ++c) {
        float4 v = sr4[c];
        v.x = __expf(v.x - rmax); v.y = __expf(v.y - rmax);
        v.z = __expf(v.z - rmax); v.w = __expf(v.w - rmax);
        sr4[c] = v;
        sm += v.x + v.y + v.z + v.w;
    }
    __syncthreads();
    red[tid] = sm; __syncthreads();
    if (sub == 0) {
        float m = 0.f;
        for (int j = 0; j < 8; ++j) m += red[row * 8 + j];
        rowstat[row] = m;
    }
    __syncthreads();
    float rinv = 1.f / rowstat[row];
    float4* wrow4 = (float4*)(w_out + ((size_t)bh * 2048 + (q0 + row)) * 2048 + sub * 256);
    for (int c = 0; c < 64; ++c) {
        float4 v = sr4[c];
        v.x *= rinv; v.y *= rinv; v.z *= rinv; v.w *= rinv;
        sr4[c] = v;            // keep normalized w in LDS for phase 3
        wrow4[c] = v;          // the 537MB output: written once, never re-read
    }
    __syncthreads();

    // ---- Phase 3: att = w @ v ; waves 0..3, rowblocks 0/1 per wave ----
    if (wv < 4) {
        int ncol = wv * 16 + ln;
        const unsigned short* vtb = vt + ((size_t)bh * 64 + ncol) * 2048;
        v8f acc[2] = {};
        for (int k0 = 0; k0 < 2048; k0 += 32) {
            Frag b;
            const unsigned short* bp = vtb + k0 + khalfB;
            b.q[0] = *(const uint4*)bp; b.q[1] = *(const uint4*)(bp + 16);
            #pragma unroll
            for (int rb = 0; rb < 2; ++rb) {
                const float* ap = scs + (size_t)(rb * 16 + ln) * 2048 + k0 + khalfA;
                float4 f0 = *(const float4*)(ap);
                float4 f1 = *(const float4*)(ap + 4);
                float4 f2 = *(const float4*)(ap + 16);
                float4 f3 = *(const float4*)(ap + 20);
                Frag a;
                a.u[0] = pack_bf16(f0.x, f0.y); a.u[1] = pack_bf16(f0.z, f0.w);
                a.u[2] = pack_bf16(f1.x, f1.y); a.u[3] = pack_bf16(f1.z, f1.w);
                a.u[4] = pack_bf16(f2.x, f2.y); a.u[5] = pack_bf16(f2.z, f2.w);
                a.u[6] = pack_bf16(f3.x, f3.y); a.u[7] = pack_bf16(f3.z, f3.w);
                acc[rb] = __builtin_amdgcn_wmma_f32_16x16x32_bf16(false, a.v, false, b.v,
                                                                  (short)0, acc[rb], false, false);
            }
        }
        int b_ = bh >> 4;
        #pragma unroll
        for (int rb = 0; rb < 2; ++rb) {
            #pragma unroll
            for (int r = 0; r < 8; ++r) {
                int srow = q0 + rb * 16 + r + hi8;
                att_bf[((size_t)b_ * 2048 + srow) * 1024 + h * 64 + ncol] = f2bf(acc[rb][r]);
            }
        }
    }
}

// ---------------------------------------------------------------------------
// Kernel 6: out GEMM (att @ Wout^T + bout) + residual gate mix, LDS-tiled
// exactly like kernel 4 (128x64 block tiles, async double-buffered staging).
// ---------------------------------------------------------------------------
__global__ void __launch_bounds__(256)
k_out_gemm(const unsigned short* __restrict__ A, const unsigned short* __restrict__ W,
           const float* __restrict__ bout, const float* __restrict__ ch,
           const float* __restrict__ xn, float* __restrict__ outmix) {
    __shared__ unsigned short As[2][128 * 32];
    __shared__ unsigned short Bs[2][64 * 32];
    int tid = threadIdx.x, lane = tid & 31, wv = tid >> 5, ln = lane & 15;
    int blk = blockIdx.x;
    int mt = blk / 16, nt = blk % 16;            // M/128=32, N/64=16
    int m0 = mt * 128, n0 = nt * 64;
    int khalfA = (lane >> 4) * 8, khalfB = (lane >> 4) * 8, hi8 = khalfA;

    unsigned aoff[2] = { (unsigned)(size_t)&As[0][0], (unsigned)(size_t)&As[1][0] };
    unsigned boff[2] = { (unsigned)(size_t)&Bs[0][0], (unsigned)(size_t)&Bs[1][0] };

    auto stage = [&](int k0, int buf) {
        #pragma unroll
        for (int c = tid; c < 512; c += 256) {
            int row = c >> 2, seg = c & 3;
            async_copy_b128(aoff[buf] + (unsigned)(row * 64 + seg * 16),
                            A + (size_t)(m0 + row) * 1024 + k0 + seg * 8);
        }
        {
            int row = tid >> 2, seg = tid & 3;
            async_copy_b128(boff[buf] + (unsigned)(tid * 16),
                            W + (size_t)(n0 + row) * 1024 + k0 + seg * 8);
        }
    };

    v8f acc[4] = {};
    stage(0, 0);
    wait_async0();
    __syncthreads();
    for (int ks = 0; ks < 32; ++ks) {
        int buf = ks & 1;
        if (ks + 1 < 32) stage((ks + 1) * 32, buf ^ 1);
        Frag a;
        const unsigned short* ap = &As[buf][(wv * 16 + ln) * 32 + khalfA];
        a.q[0] = *(const uint4*)ap; a.q[1] = *(const uint4*)(ap + 16);
        #pragma unroll
        for (int nb = 0; nb < 4; ++nb) {
            Frag b;
            const unsigned short* bp = &Bs[buf][(nb * 16 + ln) * 32 + khalfB];
            b.q[0] = *(const uint4*)bp; b.q[1] = *(const uint4*)(bp + 16);
            acc[nb] = __builtin_amdgcn_wmma_f32_16x16x32_bf16(false, a.v, false, b.v,
                                                              (short)0, acc[nb], false, false);
        }
        wait_async0();
        __syncthreads();
    }

    float g = ch[CH_G];
    int mr = m0 + wv * 16;
    #pragma unroll
    for (int nb = 0; nb < 4; ++nb) {
        int nc = n0 + nb * 16 + ln;
        float bo = bout[nc];
        #pragma unroll
        for (int r = 0; r < 8; ++r) {
            int m = mr + r + hi8;
            float val = acc[nb][r] + bo;
            outmix[(size_t)m * 1024 + nc] = val * g + xn[(size_t)m * 1024 + nc] * (1.f - g);
        }
    }
}

// ---------------------------------------------------------------------------
// Kernel 7: LayerNorm (post) -> d_out[0 .. B*S*D)
// ---------------------------------------------------------------------------
__global__ void __launch_bounds__(256)
k_ln_post(const float* __restrict__ x, const float* __restrict__ g,
          const float* __restrict__ b, float* __restrict__ out) {
    __shared__ float red[256], red2[256];
    int row = blockIdx.x, tid = threadIdx.x;
    const float* xr = x + (size_t)row * 1024;
    float v[4], s = 0.f, s2 = 0.f;
    #pragma unroll
    for (int i = 0; i < 4; ++i) { v[i] = xr[tid + 256 * i]; s += v[i]; s2 += v[i] * v[i]; }
    red[tid] = s; red2[tid] = s2; __syncthreads();
    for (int o = 128; o > 0; o >>= 1) {
        if (tid < o) { red[tid] += red[tid + o]; red2[tid] += red2[tid + o]; }
        __syncthreads();
    }
    float mean = red[0] / 1024.f;
    float rstd = rsqrtf(red2[0] / 1024.f - mean * mean + 1e-5f);
    #pragma unroll
    for (int i = 0; i < 4; ++i) {
        int c = tid + 256 * i;
        out[(size_t)row * 1024 + c] = (v[i] - mean) * rstd * g[c] + b[c];
    }
}

// ---------------------------------------------------------------------------
extern "C" void kernel_launch(void* const* d_in, const int* in_sizes, int n_in,
                              void* d_out, int out_size, void* d_ws, size_t ws_size,
                              hipStream_t stream) {
    (void)in_sizes; (void)n_in; (void)out_size; (void)ws_size;
    const float* x        = (const float*)d_in[0];
    const float* cf       = (const float*)d_in[1];
    const float* ln_pre_g = (const float*)d_in[2];
    const float* ln_pre_b = (const float*)d_in[3];
    const float* Wqkv     = (const float*)d_in[4];
    const float* bqkv     = (const float*)d_in[5];
    const float* enc_W1   = (const float*)d_in[6];
    const float* enc_b1   = (const float*)d_in[7];
    const float* enc_g1   = (const float*)d_in[8];
    const float* enc_bb1  = (const float*)d_in[9];
    const float* enc_W2   = (const float*)d_in[10];
    const float* enc_b2   = (const float*)d_in[11];
    const float* enc_g2   = (const float*)d_in[12];
    const float* enc_bb2  = (const float*)d_in[13];
    const float* Wga      = (const float*)d_in[14];
    const float* bga      = (const float*)d_in[15];
    const float* Wgv      = (const float*)d_in[16];
    const float* bgv      = (const float*)d_in[17];
    const float* Wt       = (const float*)d_in[18];
    const float* bt       = (const float*)d_in[19];
    const float* qrot     = (const float*)d_in[20];
    const float* fscales  = (const float*)d_in[21];
    const float* Wout     = (const float*)d_in[22];
    const float* bout     = (const float*)d_in[23];
    const float* ln_post_g= (const float*)d_in[24];
    const float* ln_post_b= (const float*)d_in[25];

    char* ws = (char*)d_ws;
    size_t off = 0;
    auto carve = [&](size_t bytes) -> void* {
        void* p = ws + off;
        off = (off + bytes + 255) & ~(size_t)255;
        return p;
    };
    const size_t BSD = 4096ull * 1024ull;
    float*          ch     = (float*)carve(2048 * 4);
    float*          xn_f32 = (float*)carve(BSD * 4);
    unsigned short* xn_bf  = (unsigned short*)carve(BSD * 2);
    unsigned short* wq_bf  = (unsigned short*)carve(3072ull * 1024ull * 2);
    unsigned short* wo_bf  = (unsigned short*)carve(1024ull * 1024ull * 2);
    unsigned short* qb     = (unsigned short*)carve(BSD * 2);
    unsigned short* kb     = (unsigned short*)carve(BSD * 2);
    unsigned short* vb     = (unsigned short*)carve(BSD * 2);
    unsigned short* vtb    = (unsigned short*)carve(BSD * 2);
    unsigned short* att_bf = (unsigned short*)carve(BSD * 2);
    float*          outmix = (float*)carve(BSD * 4);

    float* out_ln = (float*)d_out;             // [B,S,D]
    float* w_out  = out_ln + BSD;              // [B,H,S,S]

    k_cvt_weights<<<1024, 256, 0, stream>>>(Wqkv, Wout, wq_bf, wo_bf);
    k_chaos<<<1, 256, 0, stream>>>(cf, enc_W1, enc_b1, enc_g1, enc_bb1,
                                   enc_W2, enc_b2, enc_g2, enc_bb2,
                                   Wga, bga, Wgv, bgv, Wt, bt, qrot, fscales, ch);
    k_ln_pre<<<4096, 256, 0, stream>>>(x, ln_pre_g, ln_pre_b, xn_f32, xn_bf);
    k_qkv_gemm<<<1536, 256, 0, stream>>>(xn_bf, wq_bf, bqkv, ch, qb, kb, vb, vtb);
    k_attention<<<2048, 256, ATT_SMEM, stream>>>(qb, kb, vtb, ch, w_out, att_bf);
    k_out_gemm<<<512, 256, 0, stream>>>(att_bf, wo_bf, bout, ch, xn_f32, outmix);
    k_ln_post<<<4096, 256, 0, stream>>>(outmix, ln_post_g, ln_post_b, out_ln);
}
